// HierarchicalRelation_89945205113719
// MI455X (gfx1250) — compile-verified, matrix operand order checked by reading, not run on hardware
//
#include <hip/hip_runtime.h>
#include <hip/hip_bf16.h>
#include <stdint.h>

#define B_    512
#define N0_   16
#define F_    2048
#define OUT_  2048
#define RS_   4
#define BN_   512
#define DEPTH_ 4

typedef __bf16 bf16_t;
typedef __attribute__((ext_vector_type(16))) __bf16 v16bf;
typedef __attribute__((ext_vector_type(8)))  __bf16 v8bf;
typedef __attribute__((ext_vector_type(8)))  float  v8f;

__device__ __forceinline__ bf16_t f2bf(float f) {
  uint32_t u = __builtin_bit_cast(uint32_t, f);
  uint32_t r = u + 0x7FFFu + ((u >> 16) & 1u);   // round-to-nearest-even
  uint16_t h = (uint16_t)(r >> 16);
  return __builtin_bit_cast(bf16_t, h);
}

// ---------------------------------------------------------------------------
// Repack fp32 weight [D, K, N] (row-major) into bf16 WMMA B-fragment order:
// packed index p = kb*N*32 + n*32 + grp*16 + e,
//   k = kb*32 + (e&8 ? 16 : 0) + grp*8 + (e&7)
// so at GEMM time lane (grp, n) reads its 16 B elements as one contiguous 32B.
// ---------------------------------------------------------------------------
__global__ void pack_w(const float* __restrict__ W, bf16_t* __restrict__ P,
                       long long total, int K, int N) {
  long long i = (long long)blockIdx.x * blockDim.x + threadIdx.x;
  long long stride = (long long)gridDim.x * blockDim.x;
  long long mat = (long long)K * N;
  for (; i < total; i += stride) {
    long long d = i / mat;
    long long o = i - d * mat;
    int e = (int)(o & 15);
    long long t = o >> 4;
    int grp = (int)(t & 1); t >>= 1;
    int n  = (int)(t % N);
    long long kb = t / N;
    int k = (int)(kb * 32) + ((e & 8) ? 16 : 0) + grp * 8 + (e & 7);
    P[i] = f2bf(W[d * mat + (long long)k * N + n]);
  }
}

__global__ void relu_cvt(const float* __restrict__ in, bf16_t* __restrict__ out,
                         long long n) {
  long long i = (long long)blockIdx.x * blockDim.x + threadIdx.x;
  long long s = (long long)gridDim.x * blockDim.x;
  for (; i < n; i += s) out[i] = f2bf(fmaxf(in[i], 0.f));
}

// s[b,f] = sum_{j<w} h[b,j,f]   (h: [B, w, F] fp32) -> bf16
__global__ void rowsum_cvt(const float* __restrict__ h, bf16_t* __restrict__ s,
                           int Bb, int w, int Ff) {
  long long n = (long long)Bb * Ff;
  long long i = (long long)blockIdx.x * blockDim.x + threadIdx.x;
  long long st = (long long)gridDim.x * blockDim.x;
  for (; i < n; i += st) {
    int b = (int)(i / Ff);
    int f = (int)(i - (long long)b * Ff);
    const float* p = h + ((size_t)b * w) * Ff + f;
    float acc = 0.f;
    for (int j = 0; j < w; ++j) acc += p[(size_t)j * Ff];
    s[i] = f2bf(acc);
  }
}

__global__ void zero_f32(float* __restrict__ p, long long n) {
  long long i = (long long)blockIdx.x * blockDim.x + threadIdx.x;
  long long s = (long long)gridDim.x * blockDim.x;
  for (; i < n; i += s) p[i] = 0.f;
}

__global__ void scale_f32(float* __restrict__ p, float sc, long long n) {
  long long i = (long long)blockIdx.x * blockDim.x + threadIdx.x;
  long long s = (long long)gridDim.x * blockDim.x;
  for (; i < n; i += s) p[i] *= sc;
}

// ---------------------------------------------------------------------------
// bf16 WMMA GEMM:  C[M,N] = op(A[M,K] @ Bp + bias*bias_scale)
//  A row m starts at A + ((m/rowsPerBatch)*batchRows + m%rowsPerBatch)*rowStride
//  (overlapping sliding-window rows for the relation GEMMs; dense if
//   rowsPerBatch==M, rowStride==K).
//  OUT_MODE: 0 = fp32 store, 1 = bf16 store, 2 = fp32 accumulate (+=)
//  Requires M%128==0, N%128==0, K%32==0.
//  Block: 256 threads = 8 waves, tiled 4(M) x 2(N); wave tile 32x64
//  (2 A fragments x 4 B fragments -> 8 WMMA per K-step; B reused across
//   both A rows for ~21 FLOP/byte of L2 traffic).
// ---------------------------------------------------------------------------
template <int OUT_MODE, bool RELU>
__global__ __launch_bounds__(256)
void gemm_bf16_wmma(const bf16_t* __restrict__ A, const bf16_t* __restrict__ Bp,
                    const float* __restrict__ bias, float bias_scale,
                    void* __restrict__ Cv, int M, int N, int K,
                    int rowsPerBatch, int batchRows, int rowStride) {
  const int lane  = threadIdx.x & 31;
  const int wave  = threadIdx.x >> 5;
  const int tileM = blockIdx.y * 128 + (wave >> 1) * 32;
  const int tileN = blockIdx.x * 128 + (wave & 1) * 64;
  const int grp   = lane >> 4;     // 0: K 0-7/16-23, 1: K 8-15/24-31
  const int lm    = lane & 15;

  const bf16_t* Arow[2];
#pragma unroll
  for (int i = 0; i < 2; ++i) {
    const int mrow = tileM + i * 16 + lm;
    const int bidx = mrow / rowsPerBatch;
    const int irow = mrow - bidx * rowsPerBatch;
    Arow[i] =
        A + ((long long)bidx * batchRows + irow) * (long long)rowStride + grp * 8;
  }

  const int c0 = tileN + lm;
  const bf16_t* Bbase = Bp + (size_t)c0 * 32 + (size_t)grp * 16;

  v8f acc[2][4];
#pragma unroll
  for (int i = 0; i < 2; ++i)
#pragma unroll
    for (int j = 0; j < 4; ++j) acc[i][j] = (v8f){};

  const int kblocks = K >> 5;
  for (int kb = 0; kb < kblocks; ++kb) {
    v16bf af[2];
#pragma unroll
    for (int i = 0; i < 2; ++i) {
      const bf16_t* ap = Arow[i] + (kb << 5);
      v8bf a0 = *reinterpret_cast<const v8bf*>(ap);
      v8bf a1 = *reinterpret_cast<const v8bf*>(ap + 16);
      af[i] = __builtin_shufflevector(a0, a1, 0, 1, 2, 3, 4, 5, 6, 7, 8, 9, 10,
                                      11, 12, 13, 14, 15);
    }

    const bf16_t* bp = Bbase + (size_t)kb * (size_t)N * 32;
    v16bf bf4[4];
#pragma unroll
    for (int j = 0; j < 4; ++j)
      bf4[j] = *reinterpret_cast<const v16bf*>(bp + (size_t)j * 16 * 32);

#pragma unroll
    for (int i = 0; i < 2; ++i)
#pragma unroll
      for (int j = 0; j < 4; ++j)
        acc[i][j] = __builtin_amdgcn_wmma_f32_16x16x32_bf16(
            false, af[i], false, bf4[j], (short)0, acc[i][j], false, false);
  }

  float bv[4] = {0.f, 0.f, 0.f, 0.f};
  if (bias) {
#pragma unroll
    for (int j = 0; j < 4; ++j) bv[j] = bias[c0 + j * 16] * bias_scale;
  }

#pragma unroll
  for (int i = 0; i < 2; ++i) {
#pragma unroll
    for (int v = 0; v < 8; ++v) {
      // C/D layout: VGPR v -> M = v + 8*grp within the 16x16 tile
      const long long r = (long long)tileM + i * 16 + grp * 8 + v;
      const long long base = r * N + c0;
#pragma unroll
      for (int j = 0; j < 4; ++j) {
        float xv = acc[i][j][v] + bv[j];
        if (RELU) xv = fmaxf(xv, 0.f);
        if constexpr (OUT_MODE == 1) {
          ((bf16_t*)Cv)[base + j * 16] = f2bf(xv);
        } else if constexpr (OUT_MODE == 0) {
          ((float*)Cv)[base + j * 16] = xv;
        } else {
          ((float*)Cv)[base + j * 16] += xv;
        }
      }
    }
  }
}

// ---------------------------------------------------------------------------
static inline int nblk(long long n) {
  long long b = (n + 255) / 256;
  return (int)(b > 65535 ? 65535 : b);
}

extern "C" void kernel_launch(void* const* d_in, const int* in_sizes, int n_in,
                              void* d_out, int out_size, void* d_ws,
                              size_t ws_size, hipStream_t stream) {
  const float* x      = (const float*)d_in[0];
  const float* rel_W1 = (const float*)d_in[1];
  const float* rel_b1 = (const float*)d_in[2];
  const float* rel_W2 = (const float*)d_in[3];
  const float* rel_b2 = (const float*)d_in[4];
  const float* lin_W  = (const float*)d_in[5];
  const float* lin_b  = (const float*)d_in[6];
  const float* fin_W1 = (const float*)d_in[7];
  const float* fin_b1 = (const float*)d_in[8];
  const float* fin_W2 = (const float*)d_in[9];
  const float* fin_b2 = (const float*)d_in[10];
  float* out = (float*)d_out;

  char* ws = (char*)d_ws;
  size_t off = 0;
  auto alloc = [&](size_t bytes) {
    void* p = ws + off;
    off = (off + bytes + 255) & ~(size_t)255;
    return p;
  };
  bf16_t* pW1  = (bf16_t*)alloc((size_t)DEPTH_ * RS_ * F_ * BN_ * 2);
  bf16_t* pW2  = (bf16_t*)alloc((size_t)DEPTH_ * BN_ * F_ * 2);
  bf16_t* pLW  = (bf16_t*)alloc((size_t)DEPTH_ * F_ * OUT_ * 2);
  bf16_t* pFW1 = (bf16_t*)alloc((size_t)RS_ * F_ * BN_ * 2);
  bf16_t* pFW2 = (bf16_t*)alloc((size_t)BN_ * OUT_ * 2);
  bf16_t* aBuf = (bf16_t*)alloc((size_t)B_ * N0_ * F_ * 2);
  bf16_t* z1   = (bf16_t*)alloc((size_t)B_ * 13 * BN_ * 2);
  float*  hA   = (float*)alloc((size_t)B_ * 13 * F_ * 4);
  float*  hB   = (float*)alloc((size_t)B_ * 10 * F_ * 4);
  bf16_t* sBf  = (bf16_t*)alloc((size_t)B_ * F_ * 2);

  // zero the output accumulator
  zero_f32<<<nblk(out_size), 256, 0, stream>>>(out, (long long)out_size);

  // convert + fragment-pack all weights (bf16 set ~86MB: L2-resident after
  // first touch, so direct global fragment loads need no LDS staging)
  auto pack = [&](const float* W, bf16_t* P, int D, int K, int N) {
    long long total = (long long)D * K * N;
    pack_w<<<nblk(total), 256, 0, stream>>>(W, P, total, K, N);
  };
  pack(rel_W1, pW1, DEPTH_, RS_ * F_, BN_);
  pack(rel_W2, pW2, DEPTH_, BN_, F_);
  pack(lin_W,  pLW, DEPTH_, F_, OUT_);
  pack(fin_W1, pFW1, 1, RS_ * F_, BN_);
  pack(fin_W2, pFW2, 1, BN_, OUT_);

  const float* hcur = x;
  float* hnext = hA;
  float* hother = hB;
  int Nl = N0_;
  for (int l = 0; l < DEPTH_; ++l) {
    const int w  = Nl - RS_ + 1;       // 13, 10, 7, 4
    const int M1 = B_ * w;             // multiple of 128

    // a = relu(h) in bf16
    long long nact = (long long)B_ * Nl * F_;
    relu_cvt<<<nblk(nact), 256, 0, stream>>>(hcur, aBuf, nact);

    // z1 = relu(win @ rel_W1 + b1)  (overlapping-row A, K = RS*F = 8192)
    dim3 g1(BN_ / 128, M1 / 128);
    gemm_bf16_wmma<1, true><<<g1, 256, 0, stream>>>(
        aBuf, pW1 + (size_t)l * RS_ * F_ * BN_, rel_b1 + l * BN_, 1.f, z1, M1,
        BN_, RS_ * F_, w, Nl, F_);

    // h_next = z1 @ rel_W2 + b2  (fp32)
    dim3 g2(F_ / 128, M1 / 128);
    gemm_bf16_wmma<0, false><<<g2, 256, 0, stream>>>(
        z1, pW2 + (size_t)l * BN_ * F_, rel_b2 + l * F_, 1.f, hnext, M1, F_,
        BN_, M1, 0, BN_);

    // s = sum_w h_next  -> bf16   (folds lin_W out of the window loop)
    long long ns = (long long)B_ * F_;
    rowsum_cvt<<<nblk(ns), 256, 0, stream>>>(hnext, sBf, B_, w, F_);

    // out += s @ lin_W + w * lin_b
    dim3 g3(OUT_ / 128, B_ / 128);
    gemm_bf16_wmma<2, false><<<g3, 256, 0, stream>>>(
        sBf, pLW + (size_t)l * F_ * OUT_, lin_b + l * OUT_, (float)w, out, B_,
        OUT_, F_, B_, 0, F_);

    hcur = hnext;
    float* t = hnext; hnext = hother; hother = t;
    Nl = w;
  }

  // final relation: h is [B, 4, F]; exactly the w==1 case of the layer GEMM
  long long nact = (long long)B_ * Nl * F_;
  relu_cvt<<<nblk(nact), 256, 0, stream>>>(hcur, aBuf, nact);

  dim3 gf1(BN_ / 128, B_ / 128);
  gemm_bf16_wmma<1, true><<<gf1, 256, 0, stream>>>(
      aBuf, pFW1, fin_b1, 1.f, z1, B_, BN_, RS_ * F_, 1, Nl, F_);

  dim3 gf2(OUT_ / 128, B_ / 128);
  gemm_bf16_wmma<2, false><<<gf2, 256, 0, stream>>>(
      z1, pFW2, fin_b2, 1.f, out, B_, OUT_, BN_, B_, 0, BN_);

  // mean over the 5 stacked outputs
  scale_f32<<<nblk(out_size), 256, 0, stream>>>(out, 0.2f, (long long)out_size);
}
